// MaskedCausalAttention_39599598469345
// MI455X (gfx1250) — matrix operand order repaired and checked by reading, not compiled, not run on hardware
//
#include <hip/hip_runtime.h>
#include <hip/hip_bf16.h>

typedef __attribute__((ext_vector_type(16))) _Float16 v16h;
typedef __attribute__((ext_vector_type(8)))  _Float16 v8h;
typedef __attribute__((ext_vector_type(8)))  float    v8f;
typedef __attribute__((ext_vector_type(4)))  unsigned int v4u;
typedef __attribute__((ext_vector_type(8)))  unsigned int v8u;

#define T_TOK 2052
#define TPAD  2080          // T rounded up so all K/V tile fetches stay in-bounds
#define CDIM  1024
#define NHEAD 16
#define HDIM  64
#define BATCH 2
#define MTOT  (BATCH * T_TOK)   // 4104

// ---------------------------------------------------------------------------
// helpers
// ---------------------------------------------------------------------------
__device__ __forceinline__ v16h cat8(v8h a, v8h b) {
  return __builtin_shufflevector(a, b, 0,1,2,3,4,5,6,7,8,9,10,11,12,13,14,15);
}
__device__ __forceinline__ v16h load_frag16(const _Float16* p) {
  return cat8(*(const v8h*)(p), *(const v8h*)(p + 16));
}
// A-fragment (16x32 f16): lane holds row lane&15; K halves at klo and klo+16
__device__ __forceinline__ v16h load_fragA(const _Float16* rowbase, int kloA) {
  return cat8(*(const v8h*)(rowbase + kloA), *(const v8h*)(rowbase + kloA + 16));
}
// mask: causal OR 16-wide obs block inside each 19-token block
__device__ __forceinline__ bool is_allowed(int t, int s) {
  if (s <= t) return true;
  int bt = t / 19, bs = s / 19;
  return (bt == bs) && ((t % 19) < 16) && ((s % 19) < 16);
}

// ---------------------------------------------------------------------------
// TDM: 2D tile load  global -> LDS  (data_size = 2 bytes, up-to-2D => 2 SGPR groups)
// D# packing per CDNA5 ISA 8.3 / 8.4.
// ---------------------------------------------------------------------------
__device__ __forceinline__ void tdm_load_2d(unsigned lds_off, const void* gptr,
                                            unsigned tile_d0,        // elements / row
                                            unsigned tile_d1,        // rows
                                            unsigned long long row_stride_elems) {
  unsigned long long ga = (unsigned long long)(uintptr_t)gptr;
  v4u g0;
  g0[0] = 1u;                                           // count=1, user descriptor
  g0[1] = lds_off;                                      // lds_addr [63:32]
  g0[2] = (unsigned)ga;                                 // global_addr low
  g0[3] = (unsigned)((ga >> 32) & 0x01FFFFFFull)        // global_addr[56:32]
        | (2u << 30);                                   // type = 2 ("image")
  v8u g1;
  g1[0] = (1u << 16);                                   // data_size = 1 -> 2 bytes
  g1[1] = (0x7FFFFFFFu & 0xFFFFu) << 16;                // tensor_dim0 low16 (huge: no OOB)
  g1[2] = (0x7FFFFFFFu >> 16) | ((0x7FFFFFFFu & 0xFFFFu) << 16); // dim0 hi | dim1 lo
  g1[3] = (0x7FFFFFFFu >> 16) | (tile_d0 << 16);        // dim1 hi | tile_dim0
  g1[4] = tile_d1 & 0xFFFFu;                            // tile_dim1 (tile_dim2 = 0)
  g1[5] = (unsigned)(row_stride_elems & 0xFFFFFFFFull); // tensor_dim0_stride low
  g1[6] = (unsigned)((row_stride_elems >> 32) & 0xFFFFull); // stride hi (dim1_stride=0)
  g1[7] = 0u;
  asm volatile("tensor_load_to_lds %0, %1" :: "s"(g0), "s"(g1) : "memory");
}

// ---------------------------------------------------------------------------
// fp32 -> f16 conversion
// ---------------------------------------------------------------------------
__global__ __launch_bounds__(256) void cvt_f32_f16(const float* __restrict__ s,
                                                   _Float16* __restrict__ d, int n) {
  int i = blockIdx.x * blockDim.x + threadIdx.x;
  int stride = gridDim.x * blockDim.x;
  for (; i < n; i += stride) d[i] = (_Float16)s[i];
}

// ---------------------------------------------------------------------------
// GEMM: Y[m,n] = sum_c X[m,c] * W[n,c] + bias[n]
// 8 waves; wave tile 32(M) x 64(N) (2 A-frags x 4 B-frags = 8 WMMA / k-step),
// software-pipelined fragment loads (double buffered in registers).
// mode 0: Q -> [b,h,TPAD,64] f16   mode 1: K -> same
// mode 2: V -> [b,h,64,TPAD] f16 (transposed)   mode 3: fp32 out[m*1024+n]
// ---------------------------------------------------------------------------
__global__ __launch_bounds__(256)
void gemm_wmma(const _Float16* __restrict__ X, const _Float16* __restrict__ W,
               const float* __restrict__ bias, int mode,
               _Float16* __restrict__ outh, float* __restrict__ outf) {
  const int lane = threadIdx.x & 31;
  const int wave = threadIdx.x >> 5;
  const int wm = wave & 1, wn = wave >> 1;
  const int m0 = blockIdx.x * 64 + wm * 32;
  const int n0 = blockIdx.y * 256 + wn * 64;
  const int ln   = lane & 15;
  const int hi8  = (lane & 16) ? 8 : 0;
  const int kloA = (lane & 16) ? 8 : 0;
  const int kloB = (lane & 16) ? 8 : 0;

  // branch-free A row pointers (padding rows clamp to row 0; results discarded/masked)
  const _Float16* arow[2];
#pragma unroll
  for (int i = 0; i < 2; ++i) {
    int mr = m0 + i * 16 + ln;
    int xr;
    if (mode == 3) xr = (mr < MTOT) ? mr : 0;
    else { int b = mr / TPAD, t = mr % TPAD; xr = (t < T_TOK) ? (b * T_TOK + t) : 0; }
    arow[i] = X + (size_t)xr * CDIM;
  }
  const _Float16* wrow[4];
#pragma unroll
  for (int j = 0; j < 4; ++j) wrow[j] = W + (size_t)(n0 + j * 16 + ln) * CDIM + kloB;

  v8f acc[2][4] = {};
  v16h a0 = load_fragA(arow[0], kloA), a1 = load_fragA(arow[1], kloA);
  v16h b0 = load_frag16(wrow[0]), b1 = load_frag16(wrow[1]);
  v16h b2 = load_frag16(wrow[2]), b3 = load_frag16(wrow[3]);

  for (int k0 = 0; k0 < CDIM; k0 += 32) {
    v16h na0 = a0, na1 = a1, nb0 = b0, nb1 = b1, nb2 = b2, nb3 = b3;
    int kn = k0 + 32;
    if (kn < CDIM) {        // prefetch next k-step while this step's WMMAs run
      na0 = load_fragA(arow[0] + kn, kloA);
      na1 = load_fragA(arow[1] + kn, kloA);
      nb0 = load_frag16(wrow[0] + kn);
      nb1 = load_frag16(wrow[1] + kn);
      nb2 = load_frag16(wrow[2] + kn);
      nb3 = load_frag16(wrow[3] + kn);
    }
    acc[0][0] = __builtin_amdgcn_wmma_f32_16x16x32_f16(false, a0, false, b0, (short)0, acc[0][0], false, false);
    acc[1][0] = __builtin_amdgcn_wmma_f32_16x16x32_f16(false, a1, false, b0, (short)0, acc[1][0], false, false);
    acc[0][1] = __builtin_amdgcn_wmma_f32_16x16x32_f16(false, a0, false, b1, (short)0, acc[0][1], false, false);
    acc[1][1] = __builtin_amdgcn_wmma_f32_16x16x32_f16(false, a1, false, b1, (short)0, acc[1][1], false, false);
    acc[0][2] = __builtin_amdgcn_wmma_f32_16x16x32_f16(false, a0, false, b2, (short)0, acc[0][2], false, false);
    acc[1][2] = __builtin_amdgcn_wmma_f32_16x16x32_f16(false, a1, false, b2, (short)0, acc[1][2], false, false);
    acc[0][3] = __builtin_amdgcn_wmma_f32_16x16x32_f16(false, a0, false, b3, (short)0, acc[0][3], false, false);
    acc[1][3] = __builtin_amdgcn_wmma_f32_16x16x32_f16(false, a1, false, b3, (short)0, acc[1][3], false, false);
    a0 = na0; a1 = na1; b0 = nb0; b1 = nb1; b2 = nb2; b3 = nb3;
  }

#pragma unroll
  for (int j = 0; j < 4; ++j) {
    int n = n0 + j * 16 + ln;
    float bv = bias[n];
#pragma unroll
    for (int i = 0; i < 2; ++i) {
#pragma unroll
      for (int r = 0; r < 8; ++r) {
        int m = m0 + i * 16 + r + hi8;
        float val = acc[i][j][r] + bv;
        if (mode == 3) {
          if (m < MTOT) outf[(size_t)m * CDIM + n] = val;
        } else {
          int b = m / TPAD, t = m % TPAD;
          int hh = n >> 6, d = n & 63;
          size_t idx;
          if (mode == 2) idx = (((size_t)b * NHEAD + hh) * HDIM + d) * TPAD + t;
          else           idx = (((size_t)b * NHEAD + hh) * TPAD + t) * HDIM + d;
          outh[idx] = (_Float16)val;
        }
      }
    }
  }
}

// ---------------------------------------------------------------------------
// Flash attention, block-cooperative:
//   - 8 waves x 16-row strips = 128 t-rows per block
//   - K/V chunks (32 s-columns) staged in LDS by the Tensor Data Mover,
//     double-buffered: waves 0/1 issue tensor_load_to_lds for chunk i+1,
//     s_wait_tensorcnt + block barrier gate chunk i's compute.
//   - online softmax via shfl within 16-lane half-groups; P re-layout through
//     a per-wave LDS bounce into a WMMA A-fragment.
// ---------------------------------------------------------------------------
__global__ __launch_bounds__(256)
void flash_attn(const _Float16* __restrict__ Q, const _Float16* __restrict__ K,
                const _Float16* __restrict__ Vt, _Float16* __restrict__ att) {
  __shared__ __align__(16) _Float16 kbuf[2][32 * HDIM];   // [srel][d]
  __shared__ __align__(16) _Float16 vbuf[2][HDIM * 32];   // [d][srel]
  __shared__ __align__(16) _Float16 plds[8][16 * 32];

  const int lane = threadIdx.x & 31;
  const int wave = threadIdx.x >> 5;
  const int t0 = blockIdx.x * 128 + wave * 16;
  const bool active = (t0 < T_TOK);
  const int h = blockIdx.y, b = blockIdx.z;

  const _Float16* Qb = Q  + (((size_t)b * NHEAD + h) * TPAD) * HDIM;
  const _Float16* Kb = K  + (((size_t)b * NHEAD + h) * TPAD) * HDIM;
  const _Float16* Vb = Vt + (((size_t)b * NHEAD + h) * HDIM) * TPAD;
  _Float16* lp = plds[wave];

  const int ln   = lane & 15;
  const int hi8  = (lane & 16) ? 8 : 0;
  const int kloA = (lane & 16) ? 8 : 0;
  const int kloB = (lane & 16) ? 8 : 0;

  // per-wave Q fragments (d = 0..63)
  v16h qa0 = {}, qa1 = {};
  if (active) {
    const _Float16* rb = Qb + (size_t)(t0 + ln) * HDIM;
    qa0 = load_fragA(rb, kloA);
    qa1 = load_fragA(rb + 32, kloA);
  }

  v8f o[4] = {};
  float mrow[8], lrow[8];
#pragma unroll
  for (int r = 0; r < 8; ++r) { mrow[r] = -1e30f; lrow[r] = 0.0f; }

  // this wave's s extent
  int my_send = 0;
  if (active) {
    int th = t0 + 15;
    my_send = th + 1;
    int bd = (th / 19) * 19 + 16;
    if (bd > my_send) my_send = bd;
    my_send = (my_send + 31) & ~31;
  }
  // block s extent (from last active row)
  int lastrow = blockIdx.x * 128 + 127;
  if (lastrow > T_TOK - 1) lastrow = T_TOK - 1;
  int se_blk = lastrow + 1;
  { int bd = (lastrow / 19) * 19 + 16; if (bd > se_blk) se_blk = bd; }
  se_blk = (se_blk + 31) & ~31;

  // prologue: TDM chunk 0 into buffer 0
  if (wave == 0)
    tdm_load_2d((unsigned)(uintptr_t)(void*)&kbuf[0][0], Kb, 32 * HDIM, 1, 32 * HDIM);
  if (wave == 1)
    tdm_load_2d((unsigned)(uintptr_t)(void*)&vbuf[0][0], Vb, 32, HDIM, TPAD);

  int buf = 0;
  for (int s0 = 0; s0 < se_blk; s0 += 32, buf ^= 1) {
    const bool hasnext = (s0 + 32) < se_blk;
    if (hasnext) {   // kick chunk i+1 while chunk i computes
      if (wave == 0)
        tdm_load_2d((unsigned)(uintptr_t)(void*)&kbuf[buf ^ 1][0],
                    Kb + (size_t)(s0 + 32) * HDIM, 32 * HDIM, 1, 32 * HDIM);
      if (wave == 1)
        tdm_load_2d((unsigned)(uintptr_t)(void*)&vbuf[buf ^ 1][0],
                    Vb + (size_t)(s0 + 32), 32, HDIM, TPAD);
    }
    if (wave < 2) {
      if (hasnext) __builtin_amdgcn_s_wait_tensorcnt(1);
      else         __builtin_amdgcn_s_wait_tensorcnt(0);
    }
    __syncthreads();   // chunk i resident in LDS for all waves

    if (active && s0 < my_send) {
      const _Float16* kb = &kbuf[buf][0];
      const _Float16* vb = &vbuf[buf][0];

      // ---- S = Q K^T : two 16x16 tiles from LDS
      v8f sa = {}, sb = {};
      {
        const _Float16* kr = kb + (size_t)ln * HDIM;
        sa = __builtin_amdgcn_wmma_f32_16x16x32_f16(false, qa0, false,
              load_frag16(kr + kloB),      (short)0, sa, false, false);
        sa = __builtin_amdgcn_wmma_f32_16x16x32_f16(false, qa1, false,
              load_frag16(kr + 32 + kloB), (short)0, sa, false, false);
        const _Float16* kr2 = kb + (size_t)(16 + ln) * HDIM;
        sb = __builtin_amdgcn_wmma_f32_16x16x32_f16(false, qa0, false,
              load_frag16(kr2 + kloB),      (short)0, sb, false, false);
        sb = __builtin_amdgcn_wmma_f32_16x16x32_f16(false, qa1, false,
              load_frag16(kr2 + 32 + kloB), (short)0, sb, false, false);
      }

      // ---- mask + online softmax
      float corr[8];
#pragma unroll
      for (int r = 0; r < 8; ++r) {
        int t = t0 + r + hi8;
        int sA = s0 + ln, sB = sA + 16;
        float xa = is_allowed(t, sA) ? sa[r] * 0.125f : -1e30f;
        float xb = is_allowed(t, sB) ? sb[r] * 0.125f : -1e30f;
        float mx = fmaxf(xa, xb);
#pragma unroll
        for (int off = 1; off < 16; off <<= 1)
          mx = fmaxf(mx, __shfl_xor(mx, off, 32));
        float mn = fmaxf(mrow[r], mx);
        float c  = __expf(mrow[r] - mn);
        float pa = __expf(xa - mn);
        float pb = __expf(xb - mn);
        float ps = pa + pb;
#pragma unroll
        for (int off = 1; off < 16; off <<= 1)
          ps += __shfl_xor(ps, off, 32);
        lrow[r] = lrow[r] * c + ps;
        mrow[r] = mn;
        corr[r] = c;
        lp[(r + hi8) * 32 + ln]      = (_Float16)pa;
        lp[(r + hi8) * 32 + ln + 16] = (_Float16)pb;
      }
#pragma unroll
      for (int j = 0; j < 4; ++j)
#pragma unroll
        for (int r = 0; r < 8; ++r) o[j][r] *= corr[r];

      asm volatile("s_wait_dscnt 0" ::: "memory");   // wave-local P store->load order
      v16h pfr;
      {
        const _Float16* pr = lp + ln * 32;
        pfr = cat8(*(const v8h*)(pr + kloA), *(const v8h*)(pr + kloA + 16));
      }

      // ---- O += P * V  (V tile in LDS as [d][srel])
#pragma unroll
      for (int j = 0; j < 4; ++j) {
        const _Float16* vr = vb + (size_t)(j * 16 + ln) * 32 + kloB;
        o[j] = __builtin_amdgcn_wmma_f32_16x16x32_f16(false, pfr, false,
                 load_frag16(vr), (short)0, o[j], false, false);
      }
    }
    __syncthreads();   // all readers done before buffer is re-filled
  }

  if (active) {
#pragma unroll
    for (int j = 0; j < 4; ++j) {
#pragma unroll
      for (int r = 0; r < 8; ++r) {
        int t = t0 + r + hi8;
        if (t < T_TOK) {
          float val = o[j][r] / lrow[r];
          att[((size_t)b * T_TOK + t) * CDIM + h * HDIM + j * 16 + ln] = (_Float16)val;
        }
      }
    }
  }
}

// ---------------------------------------------------------------------------
// launch
// ---------------------------------------------------------------------------
extern "C" void kernel_launch(void* const* d_in, const int* in_sizes, int n_in,
                              void* d_out, int out_size, void* d_ws, size_t ws_size,
                              hipStream_t stream) {
  const float* x  = (const float*)d_in[0];
  const float* Wq = (const float*)d_in[1];
  const float* bq = (const float*)d_in[2];
  const float* Wk = (const float*)d_in[3];
  const float* bk = (const float*)d_in[4];
  const float* Wv = (const float*)d_in[5];
  const float* bv = (const float*)d_in[6];
  const float* Wp = (const float*)d_in[7];
  const float* bp = (const float*)d_in[8];
  float* out = (float*)d_out;
  (void)in_sizes; (void)n_in; (void)out_size; (void)ws_size;

  char* ws = (char*)d_ws;
  size_t off = 0;
  auto alloc = [&](size_t bytes) -> void* {
    void* p = ws + off;
    off += (bytes + 255) & ~(size_t)255;
    return p;
  };
  const size_t MT = (size_t)MTOT;
  _Float16* xh  = (_Float16*)alloc(MT * CDIM * 2);
  _Float16* wqh = (_Float16*)alloc((size_t)CDIM * CDIM * 2);
  _Float16* wkh = (_Float16*)alloc((size_t)CDIM * CDIM * 2);
  _Float16* wvh = (_Float16*)alloc((size_t)CDIM * CDIM * 2);
  _Float16* wph = (_Float16*)alloc((size_t)CDIM * CDIM * 2);
  _Float16* qh  = (_Float16*)alloc((size_t)BATCH * NHEAD * TPAD * HDIM * 2);
  _Float16* kh  = (_Float16*)alloc((size_t)BATCH * NHEAD * TPAD * HDIM * 2);
  _Float16* vth = (_Float16*)alloc((size_t)BATCH * NHEAD * TPAD * HDIM * 2);
  _Float16* ath = (_Float16*)alloc(MT * CDIM * 2);

  cvt_f32_f16<<<2048, 256, 0, stream>>>(x,  xh,  (int)(MT * CDIM));
  cvt_f32_f16<<<1024, 256, 0, stream>>>(Wq, wqh, CDIM * CDIM);
  cvt_f32_f16<<<1024, 256, 0, stream>>>(Wk, wkh, CDIM * CDIM);
  cvt_f32_f16<<<1024, 256, 0, stream>>>(Wv, wvh, CDIM * CDIM);
  cvt_f32_f16<<<1024, 256, 0, stream>>>(Wp, wph, CDIM * CDIM);

  // Q, K, V projections: 65 * 64 = 4160 = BATCH*TPAD rows exactly
  dim3 gqkv(65, 4, 1);
  gemm_wmma<<<gqkv, 256, 0, stream>>>(xh, wqh, bq, 0, qh,  nullptr);
  gemm_wmma<<<gqkv, 256, 0, stream>>>(xh, wkh, bk, 1, kh,  nullptr);
  gemm_wmma<<<gqkv, 256, 0, stream>>>(xh, wvh, bv, 2, vth, nullptr);

  flash_attn<<<dim3(17, NHEAD, BATCH), 256, 0, stream>>>(qh, kh, vth, ath);

  // output projection -> fp32 d_out (65 blocks cover 4160 rows, bounds-checked)
  gemm_wmma<<<dim3(65, 4, 1), 256, 0, stream>>>(ath, wph, bp, 3, nullptr, out);
}
